// ODST_30064771072292
// MI455X (gfx1250) — matrix operand order repaired and checked by reading, not compiled, not run on hardware
//
#include <hip/hip_runtime.h>

// ODST forward for MI455X (gfx1250, wave32, WMMA).
// B=4096, F=256, T=128, D=6, U=16, NB=64.
//
// Pipeline:
//  K1 prep : sparsemax over D (6-elem rows) -> selectors^T as f16 (N=T*D x K=F),
//            x -> f16, response -> f16, thresholds / exp(-log_temp) precompute.
//  K2 gemm1: fv(4096x768) = x(4096x256) @ sel(256x768) via v_wmma_f32_16x16x32_f16,
//            one wave per 16x16 tile, K=256 -> 8 WMMA per tile.
//  K3 trees: per 16-batch-row tile: for each of 128 trees build the 16x64 leaf-
//            weight matrix (bit-select outer product of sparsemoid pairs) into
//            LDS in A-fragment-friendly row-major layout, then 2 WMMA (K=64)
//            against response[t] (K-contiguous in memory), accumulating the
//            tree sum in the f32 accumulator. Final store = (4096x16) output.

#define BB    4096
#define FF    256
#define TT    128
#define DD    6
#define UU    16
#define NBINS 64
#define N1    (TT*DD)   // 768

typedef __attribute__((ext_vector_type(16))) _Float16 v16h;
typedef __attribute__((ext_vector_type(8)))  _Float16 v8h;
typedef __attribute__((ext_vector_type(8)))  float    v8f;

// ---------------------------------------------------------------- kernel 1
__global__ void __launch_bounds__(256)
odst_prep(const float* __restrict__ x,
          const float* __restrict__ logits,     // (F,T,D)
          const float* __restrict__ thr_in,     // (T,D)
          const float* __restrict__ logtemp,    // (T,D)
          const float* __restrict__ resp,       // (T,U,NB)
          _Float16* __restrict__ xh,            // (B,F) f16
          _Float16* __restrict__ selT,          // (T*D, F) f16  (B^T for gemm1)
          _Float16* __restrict__ resph,         // (T,U,NB) f16
          float* __restrict__ thr,
          float* __restrict__ invt)
{
    int i = blockIdx.x * blockDim.x + threadIdx.x;
    if (i < BB*FF)       xh[i]    = (_Float16)x[i];
    if (i < TT*UU*NBINS) resph[i] = (_Float16)resp[i];
    if (i < TT*DD) { thr[i] = thr_in[i]; invt[i] = __expf(-logtemp[i]); }
    if (i < FF*TT) {
        int f = i / TT, t = i % TT;
        const float* zp = logits + (f*TT + t)*DD;
        float z[DD], zs[DD];
        #pragma unroll
        for (int d = 0; d < DD; d++) { z[d] = zp[d]; zs[d] = z[d]; }
        // sort descending (6 elements, bubble network)
        #pragma unroll
        for (int a = 0; a < DD-1; a++)
            #pragma unroll
            for (int b = 0; b < DD-1-a; b++)
                if (zs[b] < zs[b+1]) { float tv = zs[b]; zs[b] = zs[b+1]; zs[b+1] = tv; }
        float run = 0.f, cs[DD];
        int k = 0;
        #pragma unroll
        for (int d = 0; d < DD; d++) {
            run += zs[d]; cs[d] = run;
            if (1.f + (float)(d+1)*zs[d] > cs[d]) k++;
        }
        float tau = (cs[k-1] - 1.f) / (float)k;
        #pragma unroll
        for (int d = 0; d < DD; d++) {
            float v = z[d] - tau; v = v > 0.f ? v : 0.f;
            selT[(t*DD + d)*FF + f] = (_Float16)v;   // transposed: row n, K contiguous
        }
    }
}

// ---------------------------------------------------------------- kernel 2
__global__ void __launch_bounds__(256)
odst_gemm1(const _Float16* __restrict__ xh,    // (B,F)
           const _Float16* __restrict__ selT,  // (N1,F)
           float* __restrict__ fv)             // (B,N1) f32
{
    int lane = threadIdx.x & 31;
    int wv   = threadIdx.x >> 5;
    int tile = blockIdx.x * 8 + wv;            // 12288 tiles total
    int tm   = tile / (N1/16);
    int tn   = tile % (N1/16);
    int hi   = lane >> 4, lo = lane & 15;

    v8f acc = {};
    int arow = tm*16 + lo;                     // A row (M = lane&15)
    int bn   = tn*16 + lo;                     // B column (N = lane&15)
    #pragma unroll
    for (int k0 = 0; k0 < FF; k0 += 32) {
        // A fragment: lanes 0-15 K=0..15, lanes 16-31 K=16..31 (matches B)
        v16h a = *(const v16h*)(xh + arow*FF + k0 + hi*16);
        // B fragment: lanes 0-15 K=0..15, lanes 16-31 K=16..31, K-contiguous in selT
        v16h b = *(const v16h*)(selT + bn*FF + k0 + hi*16);
        acc = __builtin_amdgcn_wmma_f32_16x16x32_f16(false, a, false, b,
                                                     (short)0, acc, false, false);
    }
    // C layout: VGPR r -> (M = r + 8*hi, N = lane&15)
    #pragma unroll
    for (int r = 0; r < 8; r++)
        fv[(tm*16 + r + hi*8)*N1 + tn*16 + lo] = acc[r];
}

// ---------------------------------------------------------------- kernel 3
__global__ void __launch_bounds__(256)
odst_trees(const float* __restrict__ fv,       // (B,N1)
           const float* __restrict__ thr,      // (T*D)
           const float* __restrict__ invt,     // (T*D)
           const _Float16* __restrict__ resph, // (T,U,NB)
           float* __restrict__ out)            // (B,U)
{
    __shared__ _Float16 wlds[8 * 16 * NBINS];  // 16 KB, 2 KB slice per wave
    int lane = threadIdx.x & 31;
    int wv   = threadIdx.x >> 5;
    _Float16* myw = wlds + wv * 16 * NBINS;
    int mtile = blockIdx.x * 8 + wv;           // 256 tiles of 16 batch rows
    int hi = lane >> 4, lo = lane & 15;
    int row16 = lane >> 1;                     // each lane builds half a row
    int half  = lane & 1;                      // leaf bins [32*half, 32*half+32)
    int grow  = mtile*16 + row16;

    v8f acc = {};
    for (int t = 0; t < TT; ++t) {
        // ---- sparsemoid pairs for this row's 6 depth levels ----
        const float* fvp = fv  + grow*N1 + t*DD;
        const float* th  = thr + t*DD;
        const float* iv  = invt + t*DD;
        float p[DD], q[DD];
        #pragma unroll
        for (int d = 0; d < DD; d++) {
            float tl = (fvp[d] - th[d]) * iv[d];
            p[d] = fminf(fmaxf(0.5f + 0.5f*tl, 0.f), 1.f);  // bit=0 factor
            q[d] = fminf(fmaxf(0.5f - 0.5f*tl, 0.f), 1.f);  // bit=1 factor
        }
        // ---- 64 leaf weights = outer product over 6 bits (tree-factored) ----
        float e01[4], e23[4], e45[2];
        #pragma unroll
        for (int a2 = 0; a2 < 4; a2++) {
            e01[a2] = ((a2&1) ? q[0] : p[0]) * ((a2&2) ? q[1] : p[1]);
            e23[a2] = ((a2&1) ? q[2] : p[2]) * ((a2&2) ? q[3] : p[3]);
        }
        float f5 = half ? q[5] : p[5];         // bit5 = half
        e45[0] = p[4]*f5; e45[1] = q[4]*f5;
        _Float16* wrow = myw + row16*NBINS + half*32;
        #pragma unroll
        for (int j = 0; j < 32; j++)
            wrow[j] = (_Float16)(e01[j&3] * e23[(j>>2)&3] * e45[(j>>4)&1]);
        __syncthreads();

        // ---- two WMMA: w(16x64) @ response[t]^T(64x16), K split 0..31 / 32..63
        const _Float16* bcol = resph + (t*UU + lo)*NBINS;  // column u = lane&15
        #pragma unroll
        for (int j = 0; j < 2; j++) {
            v16h a = *(const v16h*)(myw + lo*NBINS + j*32 + hi*16);
            v16h b = *(const v16h*)(bcol + j*32 + hi*16);
            acc = __builtin_amdgcn_wmma_f32_16x16x32_f16(false, a, false, b,
                                                         (short)0, acc, false, false);
        }
        __syncthreads();
    }
    #pragma unroll
    for (int r = 0; r < 8; r++)
        out[(mtile*16 + r + hi*8)*UU + lo] = acc[r];
}

// ---------------------------------------------------------------- launch
extern "C" void kernel_launch(void* const* d_in, const int* in_sizes, int n_in,
                              void* d_out, int out_size, void* d_ws, size_t ws_size,
                              hipStream_t stream)
{
    const float* x   = (const float*)d_in[0];
    const float* fsl = (const float*)d_in[1];
    const float* fth = (const float*)d_in[2];
    const float* ltp = (const float*)d_in[3];
    const float* rsp = (const float*)d_in[4];
    float* out = (float*)d_out;

    // workspace layout (16B-aligned sections, ~14.7 MB total)
    char* ws = (char*)d_ws;
    _Float16* xh    = (_Float16*)(ws + 0);         // 2,097,152 B
    _Float16* selT  = (_Float16*)(ws + 2097152);   //   393,216 B
    _Float16* resph = (_Float16*)(ws + 2490368);   //   262,144 B
    float*    thr   = (float*)   (ws + 2752512);   //     3,072 B
    float*    invt  = (float*)   (ws + 2755584);   //     3,072 B
    float*    fv    = (float*)   (ws + 2758656);   // 12,582,912 B

    odst_prep <<<(BB*FF + 255)/256, 256, 0, stream>>>(x, fsl, fth, ltp, rsp,
                                                      xh, selT, resph, thr, invt);
    odst_gemm1<<<((BB/16)*(N1/16))/8, 256, 0, stream>>>(xh, selT, fv);
    odst_trees<<<(BB/16)/8,           256, 0, stream>>>(fv, thr, invt, resph, out);
}
